// LSHAttention_51479478010632
// MI455X (gfx1250) — matrix operand order repaired and checked by reading, not compiled, not run on hardware
//
#include <hip/hip_runtime.h>
#include <hip/hip_bf16.h>
#include <math.h>

// ---------------- problem constants (fixed by reference) ----------------
#define S_LEN   4096
#define DMODEL  1024
#define NHEADS  16
#define FDIM    64
#define NB      64          // buckets per hash round
#define NH      2           // hash rounds
#define CHUNKS  128         // NH*NB
#define CLEN    64          // (NH*S_LEN)/CHUNKS
#define SORT_N  8192        // NH*S_LEN
#define BATCH   2
#define BH      (BATCH*NHEADS)

typedef __attribute__((ext_vector_type(16))) __bf16        v16bf;
typedef __attribute__((ext_vector_type(8)))  float         v8f;
typedef __attribute__((ext_vector_type(8)))  unsigned int  v8u;

// ---------------- small helpers ----------------
__device__ __forceinline__ unsigned short f2bf(float f) {
  unsigned int u = __builtin_bit_cast(unsigned int, f);
  unsigned int r = u + 0x7FFFu + ((u >> 16) & 1u);      // RNE
  return (unsigned short)(r >> 16);
}
__device__ __forceinline__ float bf2f(unsigned short h) {
  return __builtin_bit_cast(float, (unsigned int)h << 16);
}
__device__ __forceinline__ v8f zero_v8f() {
  v8f z;
#pragma unroll
  for (int i = 0; i < 8; ++i) z[i] = 0.0f;
  return z;
}
__device__ __forceinline__ v8f wmma_bf16(v16bf a, v16bf b, v8f c) {
  // (neg_a, A, neg_b, B, c_mod, C, reuse_a, reuse_b) -> v_wmma_f32_16x16x32_bf16
  return __builtin_amdgcn_wmma_f32_16x16x32_bf16(false, a, false, b, (short)0, c,
                                                 false, false);
}

// A fragment, 16x32 bf16 tile from row-major LDS [rows][ld] (ld even).
// Layout: lane<16 row=lane holds K=0..15; lane>=16 holds K=16..31.
__device__ __forceinline__ v16bf load_frag_a(const unsigned short* smem,
                                             int row0, int k0, int ld) {
  int lane = threadIdx.x & 31;
  const unsigned short* rowp =
      smem + (size_t)(row0 + (lane & 15)) * ld + k0 + ((lane >> 4) << 4);
  v8u u;
#pragma unroll
  for (int j = 0; j < 4; ++j) u[j]     = *(const unsigned int*)(rowp + 2 * j);
#pragma unroll
  for (int j = 0; j < 4; ++j) u[4 + j] = *(const unsigned int*)(rowp + 8 + 2 * j);
  return __builtin_bit_cast(v16bf, u);
}

// B fragment (32x16) where B[k][n] = buf[n][k] (K contiguous per n-row).
// Layout (this code's order): lane<16 col n=lane, K=0..15 (2/VGPR); lane>=16 K=16..31.
// Contiguous per-lane u32 reads -> vectorizable ds_load_b128.
__device__ __forceinline__ v16bf load_frag_b_t(const unsigned short* smem,
                                               int n0, int k0, int ld) {
  int lane = threadIdx.x & 31;
  const unsigned short* p =
      smem + (size_t)(n0 + (lane & 15)) * ld + k0 + ((lane >> 4) << 4);
  v8u u;
#pragma unroll
  for (int j = 0; j < 8; ++j) u[j] = *(const unsigned int*)(p + 2 * j);
  return __builtin_bit_cast(v16bf, u);
}

// ---------------- fp32 -> bf16 convert ----------------
__global__ __launch_bounds__(256) void f32_to_bf16_kernel(
    const float* __restrict__ in, unsigned short* __restrict__ out, int n) {
  int i = blockIdx.x * 256 + threadIdx.x;
  if (i < n) out[i] = f2bf(in[i]);
}

// ---------------- generic bf16 WMMA GEMM: C[M,N] = A[M,K] * B[K,N] ---------
// block tile 128x64, K step 32, 256 threads (8 waves); wave w owns M stripe
// w*16 and all four 16-wide N tiles. B is transposed while staging into LDS so
// every fragment load is K-contiguous (ds_load_b128-friendly).
__global__ __launch_bounds__(256) void gemm_bf16_kernel(
    const unsigned short* __restrict__ A, const unsigned short* __restrict__ B,
    void* __restrict__ C, int M, int N, int K, int out_bf16) {
  __shared__ __align__(16) unsigned short As[128 * 40];   // [m][k]
  __shared__ __align__(16) unsigned short Bst[64 * 40];   // [n][k] (transposed)
  const int tid = threadIdx.x, lane = tid & 31, wave = tid >> 5;
  const int m0 = blockIdx.x * 128, n0 = blockIdx.y * 64;

  v8f acc[4];
#pragma unroll
  for (int i = 0; i < 4; ++i) acc[i] = zero_v8f();

  const int arow = tid >> 1, ahalf = tid & 1;       // A staging: 128 rows x 2
  const int brow = tid >> 3, bsub = (tid & 7) * 8;  // B staging: 32 rows x 8

  for (int k0 = 0; k0 < K; k0 += 32) {
    {  // stage A 128x32 (coalesced 32B per thread)
      const unsigned int* g =
          (const unsigned int*)(A + (size_t)(m0 + arow) * K + k0 + ahalf * 16);
      unsigned int* s = (unsigned int*)(As + arow * 40 + ahalf * 16);
#pragma unroll
      for (int j = 0; j < 8; ++j) s[j] = g[j];
    }
    {  // stage B 32x64 row-major -> LDS transposed [n][k]
      const unsigned int* g =
          (const unsigned int*)(B + (size_t)(k0 + brow) * N + n0 + bsub);
#pragma unroll
      for (int j = 0; j < 4; ++j) {
        unsigned int u = g[j];
        Bst[(bsub + 2 * j) * 40 + brow]     = (unsigned short)(u & 0xffff);
        Bst[(bsub + 2 * j + 1) * 40 + brow] = (unsigned short)(u >> 16);
      }
    }
    if (k0 + 32 < K) {  // prefetch next K slab (global_prefetch_b8)
      __builtin_prefetch(A + (size_t)(m0 + arow) * K + k0 + 32 + ahalf * 16, 0, 1);
      __builtin_prefetch(B + (size_t)(k0 + 32 + brow) * N + n0 + bsub, 0, 1);
    }
    __syncthreads();
    v16bf a = load_frag_a(As, wave * 16, 0, 40);
#pragma unroll
    for (int nt = 0; nt < 4; ++nt) {
      v16bf b = load_frag_b_t(Bst, nt * 16, 0, 40);
      acc[nt] = wmma_bf16(a, b, acc[nt]);
    }
    __syncthreads();
  }

  // D layout: lane<16 -> rows m0w+0..7 col=lane; lane>=16 -> rows +8, col=lane-16
  const int col = lane & 15, rbase = (lane >> 4) * 8;
  if (out_bf16) {
    unsigned short* Co = (unsigned short*)C;
#pragma unroll
    for (int nt = 0; nt < 4; ++nt)
#pragma unroll
      for (int r = 0; r < 8; ++r)
        Co[(size_t)(m0 + wave * 16 + rbase + r) * N + n0 + nt * 16 + col] =
            f2bf(acc[nt][r]);
  } else {
    float* Co = (float*)C;
#pragma unroll
    for (int nt = 0; nt < 4; ++nt)
#pragma unroll
      for (int r = 0; r < 8; ++r)
        Co[(size_t)(m0 + wave * 16 + rbase + r) * N + n0 + nt * 16 + col] =
            acc[nt][r];
  }
}

// ---------------- LSH bucketing: build packed sort keys ----------------
// key = (4096*(argmax_bucket + hash*64) + t) << 13 | (hash*4096 + t)
__global__ __launch_bounds__(256) void lsh_bucket_kernel(
    const unsigned short* __restrict__ qbf, const float* __restrict__ rot,
    unsigned int* __restrict__ keys) {
  int idx = blockIdx.x * 256 + threadIdx.x;  // 0 .. BH*S-1
  if (idx >= BH * S_LEN) return;
  int bh = idx >> 12, t = idx & 4095;
  int b = bh >> 4, h = bh & 15;

  float qf[FDIM];
  const unsigned int* qp =
      (const unsigned int*)(qbf + ((size_t)(b * S_LEN + t) * NHEADS + h) * FDIM);
#pragma unroll
  for (int j = 0; j < FDIM / 2; ++j) {
    unsigned int u = qp[j];
    qf[2 * j]     = bf2f((unsigned short)(u & 0xffff));
    qf[2 * j + 1] = bf2f((unsigned short)(u >> 16));
  }

#pragma unroll
  for (int hash = 0; hash < NH; ++hash) {
    float rv[NB / 2];
    for (int i = 0; i < NB / 2; ++i) {
      float d = 0.f;
#pragma unroll
      for (int f = 0; f < FDIM; ++f)
        d += qf[f] * rot[(f * NH + hash) * (NB / 2) + i];
      rv[i] = d;
    }
    float best = -1e30f;
    int bi = 0;
    for (int i = 0; i < NB / 2; ++i)       // +rv block first (argmax tie rule)
      if (rv[i] > best) { best = rv[i]; bi = i; }
    for (int i = 0; i < NB / 2; ++i)
      if (-rv[i] > best) { best = -rv[i]; bi = i + NB / 2; }
    unsigned int bucket = (unsigned int)(bi + hash * NB);
    unsigned int tick   = (unsigned int)(hash * S_LEN + t);
    keys[(size_t)bh * SORT_N + tick] =
        ((S_LEN * bucket + (unsigned int)t) << 13) | tick;
  }
}

// ---------------- bitonic sort (keys unique => identical to stable sort) ----
__global__ __launch_bounds__(1024) void bitonic_sort_kernel(
    unsigned int* __restrict__ keys) {
  __shared__ unsigned int ls[SORT_N];
  unsigned int* g = keys + (size_t)blockIdx.x * SORT_N;
  for (int i = threadIdx.x; i < SORT_N; i += 1024) ls[i] = g[i];
  __syncthreads();
  for (int k = 2; k <= SORT_N; k <<= 1) {
    for (int j = k >> 1; j > 0; j >>= 1) {
      for (int i = threadIdx.x; i < SORT_N; i += 1024) {
        int p = i ^ j;
        if (p > i) {
          unsigned int a = ls[i], b = ls[p];
          bool sw = ((i & k) == 0) ? (a > b) : (a < b);
          if (sw) { ls[i] = b; ls[p] = a; }
        }
      }
      __syncthreads();
    }
  }
  for (int i = threadIdx.x; i < SORT_N; i += 1024) g[i] = ls[i];
}

// ---------------- chunked LSH attention (shared QK), WMMA bf16 ----------------
// one block per (b,h,chunk): 64 queries x 128 keys (self chunk + previous chunk)
#define KLD 72
#define VLD 136
__global__ __launch_bounds__(128) void lsh_attn_kernel(
    const unsigned short* __restrict__ qbf, const unsigned short* __restrict__ vbf,
    const unsigned int* __restrict__ sorted, unsigned short* __restrict__ o_un,
    float* __restrict__ lse_un) {
  __shared__ __align__(16) unsigned short ksm[128 * KLD];  // gathered Q(=K) rows
  __shared__ __align__(16) unsigned short vT[FDIM * VLD];  // V transposed [f][j]
  __shared__ __align__(16) unsigned short ps[64 * VLD];    // probs (bf16)
  __shared__ int info[128];
  __shared__ int ivals[64];

  const int bh = blockIdx.x >> 7, c = blockIdx.x & 127;
  const int b = bh >> 4, h = bh & 15;
  const int tid = threadIdx.x, lane = tid & 31, wave = tid >> 5;

  // ---- gather 128 rows (keys 0..63 = self chunk, 64..127 = previous chunk) ----
  {
    int e = tid;
    int cc = (e < 64) ? c : ((c + CHUNKS - 1) & (CHUNKS - 1));
    int r = e & 63;
    unsigned int sv = sorted[(size_t)bh * SORT_N + cc * CLEN + r];
    int i = (int)(sv & 8191u);   // ticker (undo-sort target)
    int t = i & 4095;            // token position
    info[e] = t + 1;
    if (e < 64) ivals[e] = i;
    size_t rowoff = ((size_t)(b * S_LEN + t) * NHEADS + h) * FDIM;
    const unsigned int* qp = (const unsigned int*)(qbf + rowoff);
    unsigned int* kd = (unsigned int*)(ksm + e * KLD);
#pragma unroll
    for (int j = 0; j < FDIM / 2; ++j) kd[j] = qp[j];
    const unsigned int* vp = (const unsigned int*)(vbf + rowoff);
#pragma unroll
    for (int j = 0; j < FDIM / 2; ++j) {
      unsigned int u = vp[j];
      vT[(2 * j) * VLD + e]     = (unsigned short)(u & 0xffff);
      vT[(2 * j + 1) * VLD + e] = (unsigned short)(u >> 16);
    }
  }
  __syncthreads();

  // ---- dots = (Q Kt)/8 : M=64 (wave stripes of 16), N=128, K=64 ----
  const int m0 = wave * 16;
  v8f acc[8];
#pragma unroll
  for (int i = 0; i < 8; ++i) acc[i] = zero_v8f();
  for (int k0 = 0; k0 < FDIM; k0 += 32) {
    v16bf a = load_frag_a(ksm, m0, k0, KLD);
#pragma unroll
    for (int nt = 0; nt < 8; ++nt) {
      v16bf bm = load_frag_b_t(ksm, nt * 16, k0, KLD);   // B[k][n] = K[n][k]
      acc[nt] = wmma_bf16(a, bm, acc[nt]);
    }
  }

  const int col = lane & 15, rbase = (lane >> 4) * 8;
  int infok[8];
#pragma unroll
  for (int nt = 0; nt < 8; ++nt) infok[nt] = info[nt * 16 + col];

  float sm[8], lse[8];
#pragma unroll
  for (int r = 0; r < 8; ++r) {
    int qi = info[m0 + rbase + r];
    float rowmax = -1e30f;
#pragma unroll
    for (int nt = 0; nt < 8; ++nt) {
      float d = acc[nt][r] * 0.125f;          // / sqrt(64)
      if (qi == infok[nt]) d -= 1e5f;         // self-attention exclusion
      acc[nt][r] = d;
      rowmax = fmaxf(rowmax, d);
    }
#pragma unroll
    for (int s2 = 1; s2 < 16; s2 <<= 1)
      rowmax = fmaxf(rowmax, __shfl_xor(rowmax, s2, 32));
    float rs = 0.f;
#pragma unroll
    for (int nt = 0; nt < 8; ++nt) {
      float e = __expf(acc[nt][r] - rowmax);
      acc[nt][r] = e;
      rs += e;
    }
#pragma unroll
    for (int s2 = 1; s2 < 16; s2 <<= 1) rs += __shfl_xor(rs, s2, 32);
    sm[r] = rs;
    lse[r] = rowmax + __logf(rs);
  }

  // ---- unnormalized probs -> LDS (bf16) ----
#pragma unroll
  for (int nt = 0; nt < 8; ++nt)
#pragma unroll
    for (int r = 0; r < 8; ++r)
      ps[(m0 + rbase + r) * VLD + nt * 16 + col] = f2bf(acc[nt][r]);
  __syncthreads();

  // ---- O = P V : M=64, N=64, K=128 ----
  v8f acc2[4];
#pragma unroll
  for (int i = 0; i < 4; ++i) acc2[i] = zero_v8f();
  for (int ks2 = 0; ks2 < 4; ++ks2) {
    v16bf a = load_frag_a(ps, m0, ks2 * 32, VLD);
#pragma unroll
    for (int nt = 0; nt < 4; ++nt) {
      v16bf bm = load_frag_b_t(vT, nt * 16, ks2 * 32, VLD);  // B[k][n] = vT[n][k]
      acc2[nt] = wmma_bf16(a, bm, acc2[nt]);
    }
  }

  // ---- scatter directly to UNSORTED positions (fuses undo_sort) ----
#pragma unroll
  for (int r = 0; r < 8; ++r) {
    int m = m0 + rbase + r;
    int i = ivals[m];
    float inv = 1.f / sm[r];
    size_t ob = ((size_t)bh * SORT_N + i) * FDIM;
#pragma unroll
    for (int nt = 0; nt < 4; ++nt)
      o_un[ob + nt * 16 + col] = f2bf(acc2[nt][r] * inv);
    if (col == 0) lse_un[(size_t)bh * SORT_N + i] = lse[r];
  }
}

// ---------------- combine hash rounds with softmax(lse) weights ----------------
__global__ __launch_bounds__(256) void combine_kernel(
    const unsigned short* __restrict__ o_un, const float* __restrict__ lse_un,
    unsigned short* __restrict__ attn) {
  int idx = blockIdx.x * 256 + threadIdx.x;  // BH*S
  if (idx >= BH * S_LEN) return;
  int bh = idx >> 12, t = idx & 4095;
  int b = bh >> 4, h = bh & 15;
  size_t base = (size_t)bh * SORT_N;
  float l0 = lse_un[base + t], l1 = lse_un[base + S_LEN + t];
  float m = fmaxf(l0, l1);
  float w0 = __expf(l0 - m), w1 = __expf(l1 - m);
  float inv = 1.f / (w0 + w1);
  w0 *= inv; w1 *= inv;
  const unsigned int* p0 = (const unsigned int*)(o_un + (base + t) * FDIM);
  const unsigned int* p1 = (const unsigned int*)(o_un + (base + S_LEN + t) * FDIM);
  unsigned int* po =
      (unsigned int*)(attn + ((size_t)(b * S_LEN + t) * NHEADS + h) * FDIM);
#pragma unroll
  for (int j = 0; j < FDIM / 2; ++j) {
    unsigned int a = p0[j], bb = p1[j];
    float lo = w0 * bf2f((unsigned short)(a & 0xffff)) +
               w1 * bf2f((unsigned short)(bb & 0xffff));
    float hi = w0 * bf2f((unsigned short)(a >> 16)) +
               w1 * bf2f((unsigned short)(bb >> 16));
    po[j] = (unsigned int)f2bf(lo) | ((unsigned int)f2bf(hi) << 16);
  }
}

// ---------------- host-side orchestration ----------------
extern "C" void kernel_launch(void* const* d_in, const int* in_sizes, int n_in,
                              void* d_out, int out_size, void* d_ws,
                              size_t ws_size, hipStream_t stream) {
  (void)in_sizes; (void)n_in; (void)out_size; (void)ws_size;
  const float* x   = (const float*)d_in[0];   // [B,S,D]
  const float* wq  = (const float*)d_in[1];   // [D,H,F]
  const float* wv  = (const float*)d_in[2];   // [D,H,F]
  const float* wo  = (const float*)d_in[3];   // [H,F,D]
  const float* rot = (const float*)d_in[4];   // [F,NH,NB/2]
  float* out = (float*)d_out;                 // [B,S,D]

  const int M = BATCH * S_LEN;                       // 8192
  const int NX = BATCH * S_LEN * DMODEL;             // 8388608
  const int NW = DMODEL * NHEADS * FDIM;             // 1048576
  const int NQ = BATCH * S_LEN * NHEADS * FDIM;      // 8388608

  char* ws = (char*)d_ws;
  size_t off = 0;
  auto carve = [&](size_t bytes) {
    char* p = ws + off;
    off += (bytes + 255) & ~(size_t)255;
    return p;
  };
  unsigned short* xbf    = (unsigned short*)carve((size_t)NX * 2);
  unsigned short* wqbf   = (unsigned short*)carve((size_t)NW * 2);
  unsigned short* wvbf   = (unsigned short*)carve((size_t)NW * 2);
  unsigned short* wobf   = (unsigned short*)carve((size_t)NW * 2);
  unsigned short* qbf    = (unsigned short*)carve((size_t)NQ * 2);
  unsigned short* vbf    = (unsigned short*)carve((size_t)NQ * 2);
  unsigned int*   keys   = (unsigned int*)carve((size_t)BH * SORT_N * 4);
  unsigned short* o_un   = (unsigned short*)carve((size_t)BH * SORT_N * FDIM * 2);
  float*          lse_un = (float*)carve((size_t)BH * SORT_N * 4);
  unsigned short* attnbf = (unsigned short*)carve((size_t)NQ * 2);

  // 1) fp32 -> bf16 conversions
  f32_to_bf16_kernel<<<(NX + 255) / 256, 256, 0, stream>>>(x, xbf, NX);
  f32_to_bf16_kernel<<<(NW + 255) / 256, 256, 0, stream>>>(wq, wqbf, NW);
  f32_to_bf16_kernel<<<(NW + 255) / 256, 256, 0, stream>>>(wv, wvbf, NW);
  f32_to_bf16_kernel<<<(NW + 255) / 256, 256, 0, stream>>>(wo, wobf, NW);

  // 2) q/v projections: [8192,1024] x [1024,1024] (bf16 out)
  dim3 ggrid(M / 128, (NHEADS * FDIM) / 64);
  gemm_bf16_kernel<<<ggrid, 256, 0, stream>>>(xbf, wqbf, qbf, M, NHEADS * FDIM,
                                              DMODEL, 1);
  gemm_bf16_kernel<<<ggrid, 256, 0, stream>>>(xbf, wvbf, vbf, M, NHEADS * FDIM,
                                              DMODEL, 1);

  // 3) LSH bucketing + packed sort keys
  lsh_bucket_kernel<<<(BH * S_LEN) / 256, 256, 0, stream>>>(qbf, rot, keys);

  // 4) per-(b,h) bitonic sort of 8192 keys in LDS
  bitonic_sort_kernel<<<BH, 1024, 0, stream>>>(keys);

  // 5) chunked attention, scatter to unsorted positions
  lsh_attn_kernel<<<BH * CHUNKS, 128, 0, stream>>>(qbf, vbf, keys, o_un, lse_un);

  // 6) combine hash rounds
  combine_kernel<<<(BH * S_LEN) / 256, 256, 0, stream>>>(o_un, lse_un, attnbf);

  // 7) output projection: [8192,1024] x [1024,1024] -> fp32 d_out
  dim3 ogrid(M / 128, DMODEL / 64);
  gemm_bf16_kernel<<<ogrid, 256, 0, stream>>>(attnbf, wobf, out, M, DMODEL,
                                              NHEADS * FDIM, 0);
}
